// Block_51264729645454
// MI455X (gfx1250) — matrix operand order repaired and checked by reading, not compiled, not run on hardware
//
#include <hip/hip_runtime.h>
#include <math.h>

typedef _Float16 h16;
typedef __attribute__((ext_vector_type(16))) _Float16 v16h;
typedef __attribute__((ext_vector_type(8)))  _Float16 v8h;
typedef __attribute__((ext_vector_type(8)))  float    v8f;

#define BB 8
#define SS 512
#define EE 768
#define HH 12
#define DD 64
#define FF 3072
#define MM (BB*SS)   /* 4096 rows */

// ---------------------------------------------------------------- WMMA helpers
__device__ __forceinline__ v8f wmma_f16(v16h a, v16h b, v8f c) {
  return __builtin_amdgcn_wmma_f32_16x16x32_f16(false, a, false, b, (short)0, c,
                                                false, false);
}

// A-matrix fragment (16x32 f16).  lane r = row, laneHi selects K halves:
// elements 0..7  -> K = 8*hi + e        (contiguous 16B)
// elements 8..15 -> K = 16 + 8*hi + e-8 (contiguous 16B)
__device__ __forceinline__ v16h fragA(const h16* base, int ld) {
  int lane = threadIdx.x & 31;
  int r = lane & 15, hi = lane >> 4;
  const h16* p = base + (long)r * ld + hi * 8;
  v8h lo = *(const v8h*)(p);
  v8h hh = *(const v8h*)(p + 16);
  v16h f;
#pragma unroll
  for (int i = 0; i < 8; ++i) { f[i] = lo[i]; f[i + 8] = hh[i]; }
  return f;
}

// B-matrix fragment (32x16 f16) read from an N-major buffer (row n, contiguous K):
// lane c = output column, elements 0..7 -> K = 8*hi + e, 8..15 -> K = 16 + 8*hi + e-8
__device__ __forceinline__ v16h fragB(const h16* base, int ld) {
  int lane = threadIdx.x & 31;
  int c = lane & 15, hi = lane >> 4;
  const h16* p = base + (long)c * ld + hi * 8;
  v8h lo = *(const v8h*)(p);
  v8h hh = *(const v8h*)(p + 16);
  v16h f;
#pragma unroll
  for (int i = 0; i < 8; ++i) { f[i] = lo[i]; f[i + 8] = hh[i]; }
  return f;
}

__device__ __forceinline__ float gelu_new(float x) {
  float x3 = x * x * x;
  return 0.5f * x * (1.f + tanhf(0.7978845608028654f * (x + 0.044715f * x3)));
}

// ---------------------------------------------------------------- GEMM
// C[M,N] = act( A[M,K](f16) @ WT[N,Kfull](f16, N-major) + bias (+oldC if accum) )
// block = 256 threads = 8 waves (4x2), block tile 128x128, wave tile 32x64.
__global__ __launch_bounds__(256) void k_gemm(
    const h16* __restrict__ A, const h16* __restrict__ WT,
    const float* __restrict__ bias,
    float* __restrict__ Cf, h16* __restrict__ Ch,
    int N, int K, int Kfull, int kOff, int accum, int act)
{
  int w  = threadIdx.x >> 5;
  int wm = w & 3, wn = w >> 2;
  int mBase = blockIdx.y * 128 + wm * 32;
  int nBase = blockIdx.x * 128 + wn * 64;
  int lane = threadIdx.x & 31;
  int col = lane & 15, hi = lane >> 4;

  v8f zero = {0.f,0.f,0.f,0.f,0.f,0.f,0.f,0.f};
  v8f acc[2][4] = {{zero,zero,zero,zero},{zero,zero,zero,zero}};

  for (int k0 = 0; k0 < K; k0 += 32) {
    v16h a0 = fragA(A + (long)mBase * K + k0, K);
    v16h a1 = fragA(A + (long)(mBase + 16) * K + k0, K);
    v16h b0 = fragB(WT + (long)(nBase +  0) * Kfull + kOff + k0, Kfull);
    v16h b1 = fragB(WT + (long)(nBase + 16) * Kfull + kOff + k0, Kfull);
    v16h b2 = fragB(WT + (long)(nBase + 32) * Kfull + kOff + k0, Kfull);
    v16h b3 = fragB(WT + (long)(nBase + 48) * Kfull + kOff + k0, Kfull);
    acc[0][0] = wmma_f16(a0, b0, acc[0][0]);
    acc[0][1] = wmma_f16(a0, b1, acc[0][1]);
    acc[0][2] = wmma_f16(a0, b2, acc[0][2]);
    acc[0][3] = wmma_f16(a0, b3, acc[0][3]);
    acc[1][0] = wmma_f16(a1, b0, acc[1][0]);
    acc[1][1] = wmma_f16(a1, b1, acc[1][1]);
    acc[1][2] = wmma_f16(a1, b2, acc[1][2]);
    acc[1][3] = wmma_f16(a1, b3, acc[1][3]);
  }

#pragma unroll
  for (int t = 0; t < 2; ++t)
#pragma unroll
    for (int u = 0; u < 4; ++u)
#pragma unroll
      for (int r = 0; r < 8; ++r) {
        int m = mBase + 16 * t + r + 8 * hi;
        int n = nBase + 16 * u + col;
        float v = acc[t][u][r];
        if (bias)  v += bias[n];
        if (accum) v += Cf[(long)m * N + n];
        if (act)   v = gelu_new(v);
        if (Cf) Cf[(long)m * N + n] = v;
        if (Ch) Ch[(long)m * N + n] = (h16)v;
      }
}

// ---------------------------------------------------------------- attention
// one block per (b, h, 16-query tile); Q,K in [B,H,S,D] f16, V in [B,H,D,S] f16.
__global__ __launch_bounds__(256) void k_attn(
    const h16* __restrict__ Q, const h16* __restrict__ Km,
    const h16* __restrict__ VT, h16* __restrict__ O,
    const unsigned char* __restrict__ maskEnc, int causal)
{
  __shared__ float sS[16][SS];
  __shared__ h16   sP[16][SS];
  __shared__ float sO[4][16][16];

  int qt = blockIdx.x, h = blockIdx.y, b = blockIdx.z;
  int qBase = qt * 16;
  long bh = (long)(b * HH + h);
  const h16* Qb = Q  + (bh * SS + qBase) * DD;
  const h16* Kb = Km +  bh * SS * DD;
  const h16* Vb = VT +  bh * DD * SS;

  int w = threadIdx.x >> 5;
  int lane = threadIdx.x & 31;
  int col = lane & 15, hi = lane >> 4;

  // phase 1: scores = (Q @ K^T) / 8 with masking, into LDS fp32
  v16h qa0 = fragA(Qb, DD);
  v16h qa1 = fragA(Qb + 32, DD);
  for (int j = 0; j < 4; ++j) {
    int kb = (w * 4 + j) * 16;
    v16h kf0 = fragB(Kb + (long)kb * DD, DD);
    v16h kf1 = fragB(Kb + (long)kb * DD + 32, DD);
    v8f s = {0.f,0.f,0.f,0.f,0.f,0.f,0.f,0.f};
    s = wmma_f16(qa0, kf0, s);
    s = wmma_f16(qa1, kf1, s);
#pragma unroll
    for (int r = 0; r < 8; ++r) {
      int qrow = qBase + r + 8 * hi;
      int key  = kb + col;
      float sv = s[r] * 0.125f;
      if (causal) { if (key > qrow) sv = -1e9f; }
      else        { if (maskEnc[b * SS + key]) sv = -1e4f; }
      sS[r + 8 * hi][key] = sv;
    }
  }
  __syncthreads();

  // phase 2: row softmax (16 lanes per row, shfl_xor width-16 reductions)
  {
    int row = threadIdx.x >> 4;
    int lc  = threadIdx.x & 15;
    float mx = -3.0e38f;
    for (int c = lc; c < SS; c += 16) mx = fmaxf(mx, sS[row][c]);
#pragma unroll
    for (int off = 8; off >= 1; off >>= 1) mx = fmaxf(mx, __shfl_xor(mx, off, 16));
    float sum = 0.f;
    for (int c = lc; c < SS; c += 16) sum += expf(sS[row][c] - mx);
#pragma unroll
    for (int off = 8; off >= 1; off >>= 1) sum += __shfl_xor(sum, off, 16);
    float rinv = 1.f / sum;
    for (int c = lc; c < SS; c += 16)
      sP[row][c] = (h16)(expf(sS[row][c] - mx) * rinv);
  }
  __syncthreads();

  // phase 3: O = P @ V, split-K over wave pairs, reduce through LDS
  int dt = w & 3, half = w >> 2;
  v8f acc = {0.f,0.f,0.f,0.f,0.f,0.f,0.f,0.f};
  for (int st = 0; st < 8; ++st) {
    int k0 = half * 256 + st * 32;
    v16h pa = fragA(&sP[0][0] + k0, SS);
    v16h vb = fragB(Vb + (long)(dt * 16) * SS + k0, SS);
    acc = wmma_f16(pa, vb, acc);
  }
  if (half == 1) {
#pragma unroll
    for (int r = 0; r < 8; ++r) sO[dt][r + 8 * hi][col] = acc[r];
  }
  __syncthreads();
  if (half == 0) {
#pragma unroll
    for (int r = 0; r < 8; ++r) {
      float v = acc[r] + sO[dt][r + 8 * hi][col];
      int m = b * SS + qBase + r + 8 * hi;
      int e = h * DD + dt * 16 + col;
      O[(long)m * EE + e] = (h16)v;
    }
  }
}

// ---------------------------------------------------------------- layernorm
__device__ __forceinline__ float blockSum(float v, float* sbuf) {
#pragma unroll
  for (int off = 16; off >= 1; off >>= 1) v += __shfl_down(v, off, 32);
  __syncthreads();
  if ((threadIdx.x & 31) == 0) sbuf[threadIdx.x >> 5] = v;
  __syncthreads();
  float s = 0.f;
#pragma unroll
  for (int i = 0; i < 8; ++i) s += sbuf[i];
  return s;
}

__global__ __launch_bounds__(256) void k_ln(
    const float* __restrict__ in, long bStride, long off0,
    const float* __restrict__ g, const float* __restrict__ bt,
    h16* __restrict__ out)
{
  __shared__ float sbuf[8];
  int m = blockIdx.x;
  int b = m / SS, s = m % SS;
  const float* row = in + (long)b * bStride + off0 + (long)s * EE;
  float lsum = 0.f;
  for (int e = threadIdx.x; e < EE; e += 256) lsum += row[e];
  float mean = blockSum(lsum, sbuf) * (1.0f / EE);
  float lv = 0.f;
  for (int e = threadIdx.x; e < EE; e += 256) { float d = row[e] - mean; lv += d * d; }
  float var = blockSum(lv, sbuf) * (1.0f / EE);
  float rstd = rsqrtf(var + 1e-5f);
  for (int e = threadIdx.x; e < EE; e += 256)
    out[(long)m * EE + e] = (h16)((row[e] - mean) * rstd * g[e] + bt[e]);
}

// ---------------------------------------------------------------- small kernels
__global__ void k_wt(const float* __restrict__ w, h16* __restrict__ wt, int K, int N) {
  long i = (long)blockIdx.x * 256 + threadIdx.x;
  if (i >= (long)K * N) return;
  int n = (int)(i / K), k = (int)(i % K);
  wt[i] = (h16)w[(long)k * N + n];
}

__global__ void k_cvt(const float* __restrict__ in, h16* __restrict__ out, long n) {
  long i = (long)blockIdx.x * 256 + threadIdx.x;
  if (i < n) out[i] = (h16)in[i];
}

// [M,E] f32 -> heads f16 ([B,H,S,D] or [B,H,D,S]); optional f32 present copy
__global__ void k_pack(const float* __restrict__ t, h16* __restrict__ dst,
                       float* __restrict__ present, int transposeV) {
  long i = (long)blockIdx.x * 256 + threadIdx.x;
  if (i >= (long)MM * EE) return;
  int m = (int)(i / EE), c = (int)(i % EE);
  int b = m / SS, s = m % SS;
  int h = c / DD, d = c % DD;
  float v = t[i];
  long o;
  if (transposeV) o = (((long)(b * HH + h) * DD + d) * SS + s);
  else            o = (((long)(b * HH + h) * SS + s) * DD + d);
  dst[o] = (h16)v;
  if (present) present[(((long)(b * HH + h) * SS + s) * DD + d)] = v;
}

__global__ void k_add(const float* __restrict__ x, const float* __restrict__ y,
                      float* __restrict__ o, long n) {
  long i = (long)blockIdx.x * 256 + threadIdx.x;
  if (i < n) o[i] = x[i] + y[i];
}

__global__ void k_gate(const float* __restrict__ gl, const float* __restrict__ a,
                       const float* __restrict__ e, float* __restrict__ encs,
                       const int* __restrict__ tauP, int first, long n) {
  long i = (long)blockIdx.x * 256 + threadIdx.x;
  if (i >= n) return;
  float tau = (float)tauP[0];
  float alpha = 1.f / (1.f + expf(-gl[i]));
  float lm = (alpha > tau) ? 1.f : 0.f;
  float vm = (alpha < 1.f - tau) ? 1.f : 0.f;
  float v = alpha * lm * a[i] + (1.f - alpha) * vm * e[i];
  encs[i] = first ? v : (encs[i] + v);
}

__global__ void k_aq(const float* __restrict__ encs, const float* __restrict__ mq,
                     float* __restrict__ aq, long n) {
  long i = (long)blockIdx.x * 256 + threadIdx.x;
  if (i >= n) return;
  int m = (int)(i / EE);
  aq[i] = encs[i] * 0.5773502691896258f * mq[m];
}

__global__ void k_out(const float* __restrict__ aq, const float* __restrict__ mf,
                      const float* __restrict__ mq, float* __restrict__ out, long n) {
  long i = (long)blockIdx.x * 256 + threadIdx.x;
  if (i >= n) return;
  int m = (int)(i / EE);
  out[i] = (aq[i] + mf[i]) * mq[m];
}

// ---------------------------------------------------------------- launcher
extern "C" void kernel_launch(void* const* d_in, const int* in_sizes, int n_in,
                              void* d_out, int out_size, void* d_ws, size_t ws_size,
                              hipStream_t stream) {
  const float* x    = (const float*)d_in[0];
  const float* enc  = (const float*)d_in[1];
  const float* mq   = (const float*)d_in[2];
  const unsigned char* maskEnc = (const unsigned char*)d_in[4];
  const int*   tau  = (const int*)d_in[6];
  const float* Wq = (const float*)d_in[7],  *Wk = (const float*)d_in[8];
  const float* Wv = (const float*)d_in[9],  *Wo = (const float*)d_in[10];
  const float* bo = (const float*)d_in[11];
  const float* fcq_w = (const float*)d_in[12], *fcq_b = (const float*)d_in[13];
  const float* fck_w = (const float*)d_in[14], *fck_b = (const float*)d_in[15];
  const float* fcv_w = (const float*)d_in[16], *fcv_b = (const float*)d_in[17];
  const float* cpr_w = (const float*)d_in[18], *cpr_b = (const float*)d_in[19];
  const float* ln1g = (const float*)d_in[20], *ln1b = (const float*)d_in[21];
  const float* ln2g = (const float*)d_in[22], *ln2b = (const float*)d_in[23];
  const float* cfc_w = (const float*)d_in[24], *cfc_b = (const float*)d_in[25];
  const float* cp2_w = (const float*)d_in[26], *cp2_b = (const float*)d_in[27];
  const float* aw[3] = {(const float*)d_in[28], (const float*)d_in[30], (const float*)d_in[32]};
  const float* ab[3] = {(const float*)d_in[29], (const float*)d_in[31], (const float*)d_in[33]};

  float* out_f = (float*)d_out;
  float* pres0 = out_f + (long)MM * EE;                 // present[0] = K
  float* pres1 = pres0 + (long)BB * HH * SS * DD;       // present[1] = V

  const long ME  = (long)MM * EE;
  const long MF  = (long)MM * FF;
  const long E2  = (long)EE * EE;
  const long E2E = (long)2 * EE * EE;
  const long EF  = (long)EE * FF;

  size_t off = 0;
  auto alloc = [&](size_t bytes) -> void* {
    void* p = (char*)d_ws + off;
    off += (bytes + 255) & ~(size_t)255;
    return p;
  };
  h16* wqT = (h16*)alloc(E2 * 2);   h16* wkT = (h16*)alloc(E2 * 2);
  h16* wvT = (h16*)alloc(E2 * 2);   h16* woT = (h16*)alloc(E2 * 2);
  h16* fcqT = (h16*)alloc(E2 * 2);  h16* fckT = (h16*)alloc(E2 * 2);
  h16* fcvT = (h16*)alloc(E2 * 2);  h16* cprT = (h16*)alloc(E2 * 2);
  h16* awT[3]; for (int i = 0; i < 3; ++i) awT[i] = (h16*)alloc(E2E * 2);
  h16* cfcT = (h16*)alloc(EF * 2);  h16* cp2T = (h16*)alloc(EF * 2);
  h16* xl   = (h16*)alloc(ME * 2);
  float* tmp = (float*)alloc(ME * 4);
  h16* qB  = (h16*)alloc(ME * 2);  h16* kB  = (h16*)alloc(ME * 2);
  h16* vTB = (h16*)alloc(ME * 2);  h16* atB = (h16*)alloc(ME * 2);
  float* aF = (float*)alloc(ME * 4);
  h16* aB  = (h16*)alloc(ME * 2);  h16* alB = (h16*)alloc(ME * 2);
  h16* elB = (h16*)alloc(ME * 2);
  h16* q2B = (h16*)alloc(ME * 2);  h16* kxB = (h16*)alloc(ME * 2);
  h16* vxT = (h16*)alloc(ME * 2);
  float* eF = (float*)alloc(ME * 4);
  h16* eB  = (h16*)alloc(ME * 2);
  float* gl   = (float*)alloc(ME * 4);
  float* encs = (float*)alloc(ME * 4);
  float* aq   = (float*)alloc(ME * 4);
  h16* hB   = (h16*)alloc(ME * 2);
  h16* midB = (h16*)alloc(MF * 2);
  float* mF = (float*)alloc(ME * 4);

  const unsigned EG = (unsigned)((ME + 255) / 256);

  auto WTr = [&](const float* w, h16* wt, int K, int N) {
    long n = (long)K * N;
    k_wt<<<(unsigned)((n + 255) / 256), 256, 0, stream>>>(w, wt, K, N);
  };
  auto GEMM = [&](const h16* A, const h16* WT, const float* bias, float* Cf, h16* Ch,
                  int N, int K, int Kfull, int kOff, int accum, int act) {
    dim3 g((unsigned)(N / 128), (unsigned)(MM / 128));
    k_gemm<<<g, 256, 0, stream>>>(A, WT, bias, Cf, Ch, N, K, Kfull, kOff, accum, act);
  };

  // weight conversions (fp32 [K,N] -> f16 [N,K])
  WTr(Wq, wqT, EE, EE); WTr(Wk, wkT, EE, EE); WTr(Wv, wvT, EE, EE); WTr(Wo, woT, EE, EE);
  WTr(fcq_w, fcqT, EE, EE); WTr(fck_w, fckT, EE, EE);
  WTr(fcv_w, fcvT, EE, EE); WTr(cpr_w, cprT, EE, EE);
  for (int i = 0; i < 3; ++i) WTr(aw[i], awT[i], 2 * EE, EE);
  WTr(cfc_w, cfcT, EE, FF); WTr(cp2_w, cp2T, FF, EE);

  // self-attention
  k_ln<<<MM, 256, 0, stream>>>(x, (long)SS * EE, 0L, ln1g, ln1b, xl);
  GEMM(xl, wqT, nullptr, tmp, nullptr, EE, EE, EE, 0, 0, 0);
  k_pack<<<EG, 256, 0, stream>>>(tmp, qB, nullptr, 0);
  GEMM(xl, wkT, nullptr, tmp, nullptr, EE, EE, EE, 0, 0, 0);
  k_pack<<<EG, 256, 0, stream>>>(tmp, kB, pres0, 0);
  GEMM(xl, wvT, nullptr, tmp, nullptr, EE, EE, EE, 0, 0, 0);
  k_pack<<<EG, 256, 0, stream>>>(tmp, vTB, pres1, 1);
  k_attn<<<dim3(SS / 16, HH, BB), 256, 0, stream>>>(qB, kB, vTB, atB, nullptr, 1);
  GEMM(atB, woT, bo, tmp, nullptr, EE, EE, EE, 0, 0, 0);
  k_add<<<EG, 256, 0, stream>>>(x, tmp, aF, ME);
  k_cvt<<<EG, 256, 0, stream>>>(aF, aB, ME);
  k_ln<<<MM, 256, 0, stream>>>(aF, (long)SS * EE, 0L, ln1g, ln1b, alB);

  // shared cross-attention query
  GEMM(alB, fcqT, fcq_b, tmp, nullptr, EE, EE, EE, 0, 0, 0);
  k_pack<<<EG, 256, 0, stream>>>(tmp, q2B, nullptr, 0);

  // three cross attentions + gates
  for (int idx = 0; idx < 3; ++idx) {
    k_ln<<<MM, 256, 0, stream>>>(enc, (long)3 * SS * EE, (long)idx * SS * EE,
                                 ln1g, ln1b, elB);
    GEMM(elB, fckT, fck_b, tmp, nullptr, EE, EE, EE, 0, 0, 0);
    k_pack<<<EG, 256, 0, stream>>>(tmp, kxB, nullptr, 0);
    GEMM(elB, fcvT, fcv_b, tmp, nullptr, EE, EE, EE, 0, 0, 0);
    k_pack<<<EG, 256, 0, stream>>>(tmp, vxT, nullptr, 1);
    k_attn<<<dim3(SS / 16, HH, BB), 256, 0, stream>>>(q2B, kxB, vxT, atB, maskEnc, 0);
    GEMM(atB, cprT, cpr_b, eF, eB, EE, EE, EE, 0, 0, 0);
    // gate logits: concat([a,e]) @ aw + ab  == a @ aw[:E] + e @ aw[E:] + ab
    GEMM(aB, awT[idx], ab[idx], gl, nullptr, EE, EE, 2 * EE, 0,   0, 0);
    GEMM(eB, awT[idx], nullptr, gl, nullptr, EE, EE, 2 * EE, EE,  1, 0);
    k_gate<<<EG, 256, 0, stream>>>(gl, aF, eF, encs, tau, idx == 0 ? 1 : 0, ME);
  }

  // aq = encs / sqrt(3) * mask_queries ; MLP ; output
  k_aq<<<EG, 256, 0, stream>>>(encs, mq, aq, ME);
  k_ln<<<MM, 256, 0, stream>>>(aq, (long)SS * EE, 0L, ln2g, ln2b, hB);
  GEMM(hB, cfcT, cfc_b, nullptr, midB, FF, EE, EE, 0, 0, 1);
  GEMM(midB, cp2T, cp2_b, mF, nullptr, EE, FF, FF, 0, 0, 0);
  k_out<<<EG, 256, 0, stream>>>(aq, mF, mq, out_f, ME);

  (void)in_sizes; (void)n_in; (void)out_size; (void)ws_size;
}
